// ReversibleQwen3CandidateAttention_24240795419156
// MI455X (gfx1250) — matrix operand order repaired and checked by reading, not compiled, not run on hardware
//
#include <hip/hip_runtime.h>
#include <hip/hip_bf16.h>

// ---------------------------------------------------------------------------
// Problem constants (match reference)
// ---------------------------------------------------------------------------
#define S_LEN   2048
#define D_MODEL 2048
#define NH      16
#define NKV     8
#define HD      128
#define TOPK    40
#define EPS_F   1e-6f
#define SCALE_F 0.08838834764831845f   // 128^-0.5
#define NEG_FILL -1000000000.0f

typedef __bf16 bf16_t;
typedef __attribute__((ext_vector_type(16))) __bf16 v16bf;
typedef __attribute__((ext_vector_type(8)))  __bf16 v8bf;
typedef __attribute__((ext_vector_type(8)))  float  v8f;
typedef __attribute__((ext_vector_type(4)))  unsigned int v4u;
typedef __attribute__((ext_vector_type(8)))  unsigned int v8u;

union V16U { v16bf v; v8bf h[2]; };

// ---------------------------------------------------------------------------
// CDNA5 async / tensor data movement helpers (inline asm, per ISA ch.10/7)
// ---------------------------------------------------------------------------
// Per-lane async 16B copy global -> LDS (bypasses VGPR data path, ASYNCcnt).
__device__ inline void async_copy_b128(unsigned lds_off, const void* gaddr) {
    asm volatile("global_load_async_to_lds_b128 %0, %1, off"
                 :: "v"(lds_off), "v"((unsigned long long)(uintptr_t)gaddr)
                 : "memory");
}
__device__ inline void wait_async0() {
    asm volatile("s_wait_asynccnt 0x0" ::: "memory");
}

// ---------------------------------------------------------------------------
// WMMA fragment helpers (layouts per ISA 7.12.2)
// ---------------------------------------------------------------------------
// A-matrix (16x32 bf16): lane = 16*hf + m holds row m.
//   elements 0..7  <-> K = k0 + hf*8 + e ; 8..15 <-> K = k0 + 16 + hf*8 + (e-8)
__device__ inline v16bf load_a_frag(const bf16_t* base, int ld, int m0, int k0) {
    const int lane = threadIdx.x & 31;
    const int hf = lane >> 4;
    const int m = lane & 15;
    const bf16_t* row = base + (size_t)(m0 + m) * ld;
    V16U u;
    u.h[0] = *(const v8bf*)(row + k0 + hf * 8);
    u.h[1] = *(const v8bf*)(row + k0 + 16 + hf * 8);
    return u.v;
}

// B-matrix (32x16 bf16) from column-contiguous storage (column n of B is a
// contiguous run along K). lane = 16*hf + n,
//   elements 0..7  <-> K = k0 + hf*8 + e ; 8..15 <-> K = k0 + 16 + hf*8 + (e-8)
__device__ inline v16bf load_b_frag(const bf16_t* colbase, size_t ld, int n0, int k0) {
    const int lane = threadIdx.x & 31;
    const int hf = lane >> 4;
    const int n = lane & 15;
    const bf16_t* col = colbase + (size_t)(n0 + n) * ld;
    V16U u;
    u.h[0] = *(const v8bf*)(col + k0 + hf * 8);
    u.h[1] = *(const v8bf*)(col + k0 + 16 + hf * 8);
    return u.v;
}

__device__ inline v8f wmma_bf16(v16bf a, v16bf b, v8f c) {
    return __builtin_amdgcn_wmma_f32_16x16x32_bf16(
        false, a, false, b, (short)0, c, false, false);
}

__device__ inline unsigned f2ord(float f) {
    unsigned u = __float_as_uint(f);
    return (u & 0x80000000u) ? ~u : (u | 0x80000000u);
}
__device__ inline float wave_sum(float v) {
    #pragma unroll
    for (int m = 16; m > 0; m >>= 1) v += __shfl_xor(v, m);
    return v;
}
__device__ inline float wave_max(float v) {
    #pragma unroll
    for (int m = 16; m > 0; m >>= 1) v = fmaxf(v, __shfl_xor(v, m));
    return v;
}
__device__ inline unsigned wave_sum_u(unsigned v) {
    #pragma unroll
    for (int m = 16; m > 0; m >>= 1) v += __shfl_xor(v, m);
    return v;
}

// ---------------------------------------------------------------------------
// Kernel 1: fp32 -> bf16 conversion
// ---------------------------------------------------------------------------
__global__ void cvt_bf16_kernel(const float* __restrict__ in,
                                bf16_t* __restrict__ out, int n) {
    int i = blockIdx.x * blockDim.x + threadIdx.x;
    int stride = gridDim.x * blockDim.x;
    for (; i < n; i += stride) out[i] = (bf16_t)in[i];
}

// ---------------------------------------------------------------------------
// Kernel 2: tiled bf16 WMMA GEMM:  C[M,N] fp32 = A[M,K] * W[N,K]^T
// Block = 256 threads (8 waves), tile 128(M) x 64(N), K-chunk 64,
// double-buffered LDS filled by async global->LDS DMA.
// Wave w: m-subtile = (w&3)*32, n-subtile = (w>>2)*32; 32x32 out per wave.
// ---------------------------------------------------------------------------
#define GBM 128
#define GBN 64
#define GBK 64

__device__ inline void stage_tiles_async(const bf16_t* A, const bf16_t* W, int K,
                                         int m0, int n0, int k0,
                                         bf16_t* As, bf16_t* Bs) {
    const int tid = threadIdx.x;
    // A tile: 128 rows x 64 elems -> 1024 chunks of 8 elems; 4 per thread
    #pragma unroll
    for (int i = 0; i < 4; ++i) {
        int c = tid * 4 + i;
        int row = c >> 3, cc = (c & 7) * 8;
        async_copy_b128((unsigned)(uintptr_t)(As + row * GBK + cc),
                        A + (size_t)(m0 + row) * K + k0 + cc);
    }
    // B tile: 64 cols x 64 elems -> 512 chunks; 2 per thread
    #pragma unroll
    for (int i = 0; i < 2; ++i) {
        int c = tid * 2 + i;
        int row = c >> 3, cc = (c & 7) * 8;
        async_copy_b128((unsigned)(uintptr_t)(Bs + row * GBK + cc),
                        W + (size_t)(n0 + row) * K + k0 + cc);
    }
}

__global__ __launch_bounds__(256)
void gemm_bf16_kernel(const bf16_t* __restrict__ A,
                      const bf16_t* __restrict__ W,
                      float* __restrict__ C,
                      int M, int N, int K) {
    __shared__ bf16_t As[2][GBM * GBK];
    __shared__ bf16_t Bs[2][GBN * GBK];

    const int tid  = threadIdx.x;
    const int wave = tid >> 5;
    const int lane = tid & 31;
    const int hf   = lane >> 4;
    const int wm   = (wave & 3) * 32;
    const int wn   = (wave >> 2) * 32;
    const int m0   = blockIdx.y * GBM;
    const int n0   = blockIdx.x * GBN;

    v8f acc00 = {}, acc01 = {}, acc10 = {}, acc11 = {};

    stage_tiles_async(A, W, K, m0, n0, 0, As[0], Bs[0]);
    wait_async0();
    __syncthreads();

    int buf = 0;
    for (int k0 = 0; k0 < K; k0 += GBK) {
        if (k0 + GBK < K)
            stage_tiles_async(A, W, K, m0, n0, k0 + GBK, As[buf ^ 1], Bs[buf ^ 1]);

        const bf16_t* as = As[buf];
        const bf16_t* bs = Bs[buf];
        #pragma unroll
        for (int kk = 0; kk < GBK; kk += 32) {
            v16bf b0 = load_b_frag(bs, GBK, wn, kk);
            v16bf b1 = load_b_frag(bs, GBK, wn + 16, kk);
            v16bf a0 = load_a_frag(as, GBK, wm, kk);
            v16bf a1 = load_a_frag(as, GBK, wm + 16, kk);
            acc00 = wmma_bf16(a0, b0, acc00);
            acc01 = wmma_bf16(a0, b1, acc01);
            acc10 = wmma_bf16(a1, b0, acc10);
            acc11 = wmma_bf16(a1, b1, acc11);
        }
        wait_async0();
        __syncthreads();
        buf ^= 1;
    }

    // Epilogue: C/D layout — lane n = lane&15, element r <-> M = hf*8 + r
    const int n = lane & 15;
    #pragma unroll
    for (int r = 0; r < 8; ++r) {
        const size_t rm0 = (size_t)(m0 + wm + hf * 8 + r) * N + n0 + wn;
        const size_t rm1 = (size_t)(m0 + wm + 16 + hf * 8 + r) * N + n0 + wn;
        C[rm0 + n]      = acc00[r];
        C[rm0 + 16 + n] = acc01[r];
        C[rm1 + n]      = acc10[r];
        C[rm1 + 16 + n] = acc11[r];
    }
}

// ---------------------------------------------------------------------------
// Kernel 3: fused RMS-norm + RoPE, fp32 in -> bf16 out.
// ---------------------------------------------------------------------------
__global__ __launch_bounds__(128)
void rmsnorm_rope_kernel(const float* __restrict__ in,
                         const float* __restrict__ cosT,
                         const float* __restrict__ sinT,
                         const float* __restrict__ w,
                         bf16_t* __restrict__ out, int nheads) {
    __shared__ float xs[HD];
    __shared__ float partial[4];

    const int s  = blockIdx.x / nheads;
    const int hh = blockIdx.x % nheads;
    const int i  = threadIdx.x;

    const size_t base = (size_t)s * nheads * HD + (size_t)hh * HD;
    float v = in[base + i];

    float ss = wave_sum(v * v);
    if ((i & 31) == 0) partial[i >> 5] = ss;
    __syncthreads();
    float var = (partial[0] + partial[1] + partial[2] + partial[3]) * (1.0f / HD);
    float inv = rsqrtf(var + EPS_F);
    float xn = v * inv * w[i];

    xs[i] = xn;
    __syncthreads();
    float other = (i < HD / 2) ? -xs[i + HD / 2] : xs[i - HD / 2];
    float c = cosT[(size_t)s * HD + i];
    float sn = sinT[(size_t)s * HD + i];
    out[base + i] = (bf16_t)(xn * c + other * sn);
}

// ---------------------------------------------------------------------------
// Kernel 4: V transpose:  vf[S, NKV*HD] fp32 -> vT[NKV][HD][S] bf16
// ---------------------------------------------------------------------------
__global__ void vtrans_kernel(const float* __restrict__ vf,
                              bf16_t* __restrict__ vT) {
    int i = blockIdx.x * blockDim.x + threadIdx.x;
    int total = NKV * HD * S_LEN;
    int stride = gridDim.x * blockDim.x;
    for (; i < total; i += stride) {
        int s = i & (S_LEN - 1);
        int d = (i >> 11) & (HD - 1);
        int hh = i >> 18;
        vT[i] = (bf16_t)vf[(size_t)s * (NKV * HD) + hh * HD + d];
    }
}

// ---------------------------------------------------------------------------
// Kernel 5: attention. Grid = (S/16, NH), 256 threads (8 waves).
// Dynamic LDS: sc[16][2048] f32 | pb[16][2048] bf16 | qs[16][128] bf16
// Q tile staged by the Tensor Data Mover (tensor_load_to_lds).
// ---------------------------------------------------------------------------
#define SC_BYTES (16 * S_LEN * 4)
#define PB_BYTES (16 * S_LEN * 2)
#define QS_BYTES (16 * HD * 2)
#define ATTN_LDS (SC_BYTES + PB_BYTES + QS_BYTES)

__global__ __launch_bounds__(256)
void attn_kernel(const bf16_t* __restrict__ qb,   // [S, NH*HD] post-rope
                 const bf16_t* __restrict__ kb,   // [S, NKV*HD] post-rope
                 const bf16_t* __restrict__ vT,   // [NKV][HD][S]
                 bf16_t* __restrict__ attn) {     // [S, NH*HD]
    extern __shared__ char smem[];
    float*  sc = (float*)smem;
    bf16_t* pb = (bf16_t*)(smem + SC_BYTES);
    bf16_t* qs = (bf16_t*)(smem + SC_BYTES + PB_BYTES);

    const int q0   = blockIdx.x * 16;
    const int h    = blockIdx.y;
    const int hkv  = h >> 1;                     // rep = NH/NKV = 2
    const int tid  = threadIdx.x;
    const int wave = tid >> 5;
    const int lane = tid & 31;
    const int hf   = lane >> 4;

    // ---- stage Q tile (2D tile: 128 elems x 16 rows, row stride 2048) via TDM
    if (wave == 0) {
        unsigned long long ga =
            (unsigned long long)(uintptr_t)(qb + (size_t)q0 * (NH * HD) + h * HD);
        unsigned lds = (unsigned)(uintptr_t)qs;
        // D# group0: count=1 | lds_addr | global_addr[56:0] | type=2
        v4u g0 = { 1u, lds, (unsigned)ga,
                   (unsigned)((ga >> 32) & 0x01FFFFFFu) | (2u << 30) };
        // D# group1: data_size=2B; tensor_dim0=2048, tensor_dim1=16;
        //            tile 128 x 16; tensor_dim0_stride=2048 elems
        v8u g1 = { 0x00010000u,                      // mask=0, data_size=1 (2B)
                   (2048u & 0xFFFFu) << 16,          // tensor_dim0[15:0]
                   (2048u >> 16) | (16u << 16),      // td0[31:16] | td1[15:0]
                   (128u << 16),                     // td1[31:16] | tile_dim0
                   16u,                              // tile_dim1 | tile_dim2=0
                   2048u,                            // dim0_stride[31:0]
                   0u, 0u };                         // stride hi, dim1_stride
        asm volatile("tensor_load_to_lds %0, %1" :: "s"(g0), "s"(g1) : "memory");
        __builtin_amdgcn_s_wait_tensorcnt(0);
    }
    __syncthreads();

    // ---- phase 1: scores = scale * Q Kh^T  (each wave: 256 keys) ----
    v16bf aq[4];
    #pragma unroll
    for (int d0 = 0; d0 < 4; ++d0)
        aq[d0] = load_a_frag(qs, HD, 0, d0 * 32);

    const bf16_t* kcol = kb + hkv * HD;
    for (int nt = 0; nt < 16; ++nt) {
        const int key0 = wave * 256 + nt * 16;
        v8f acc = {};
        #pragma unroll
        for (int d0 = 0; d0 < 4; ++d0) {
            v16bf b = load_b_frag(kcol, (size_t)(NKV * HD), key0, d0 * 32);
            acc = wmma_bf16(aq[d0], b, acc);
        }
        const int n = lane & 15;
        #pragma unroll
        for (int r = 0; r < 8; ++r)
            sc[(hf * 8 + r) * S_LEN + key0 + n] = acc[r] * SCALE_F;
    }
    __syncthreads();

    // ---- phase 2: per-row exact top-40 threshold + masked softmax ----
    for (int rr = 0; rr < 2; ++rr) {
        const int row = wave * 2 + rr;
        float* srow = sc + row * S_LEN;

        unsigned lo = 0u, hi = 0xFFFFFFFFu;
        #pragma unroll 1
        for (int it = 0; it < 32 && lo < hi; ++it) {
            unsigned mid = lo + ((hi - lo + 1) >> 1);
            unsigned cnt = 0;
            for (int j = lane; j < S_LEN; j += 32)
                cnt += (f2ord(srow[j]) >= mid) ? 1u : 0u;
            cnt = wave_sum_u(cnt);
            if (cnt >= TOPK) lo = mid; else hi = mid - 1;
        }
        const unsigned thr = lo;

        float rmax = -3.0e38f;
        for (int j = lane; j < S_LEN; j += 32) {
            float v = srow[j];
            bool keep = (f2ord(v) >= thr) || (j == q0 + row);
            v = keep ? v : NEG_FILL;
            srow[j] = v;
            rmax = fmaxf(rmax, v);
        }
        rmax = wave_max(rmax);

        float rsum = 0.0f;
        for (int j = lane; j < S_LEN; j += 32) {
            float e = __expf(srow[j] - rmax);
            srow[j] = e;
            rsum += e;
        }
        rsum = wave_sum(rsum);
        float rinv = 1.0f / rsum;

        bf16_t* prow = pb + row * S_LEN;
        for (int j = lane; j < S_LEN; j += 32)
            prow[j] = (bf16_t)(srow[j] * rinv);
    }
    __syncthreads();

    // ---- phase 3: out(16x128) = probs(16x2048) @ Vh(2048x128) ----
    const int d0w = wave * 16;
    v8f acc = {};
    const bf16_t* vcol = vT + (size_t)hkv * HD * S_LEN;
    for (int k0 = 0; k0 < S_LEN; k0 += 32) {
        v16bf a = load_a_frag(pb, S_LEN, 0, k0);
        v16bf b = load_b_frag(vcol, (size_t)S_LEN, d0w, k0);
        acc = wmma_bf16(a, b, acc);
    }
    const int n = lane & 15;
    #pragma unroll
    for (int r = 0; r < 8; ++r)
        attn[(size_t)(q0 + hf * 8 + r) * (NH * HD) + h * HD + d0w + n] = (bf16_t)acc[r];
}

// ---------------------------------------------------------------------------
// Launcher
// ---------------------------------------------------------------------------
extern "C" void kernel_launch(void* const* d_in, const int* in_sizes, int n_in,
                              void* d_out, int out_size, void* d_ws, size_t ws_size,
                              hipStream_t stream) {
    const float* x    = (const float*)d_in[0];
    const float* cosT = (const float*)d_in[1];
    const float* sinT = (const float*)d_in[2];
    const float* Wq   = (const float*)d_in[3];
    const float* Wk   = (const float*)d_in[4];
    const float* Wv   = (const float*)d_in[5];
    const float* Wo   = (const float*)d_in[6];
    const float* qnw  = (const float*)d_in[7];
    const float* knw  = (const float*)d_in[8];
    float* out = (float*)d_out;

    char* ws = (char*)d_ws;
    size_t off = 0;
    auto carve = [&](size_t bytes) { char* p = ws + off; off += (bytes + 255) & ~size_t(255); return p; };
    bf16_t* xb  = (bf16_t*)carve((size_t)S_LEN * D_MODEL * 2);
    bf16_t* wqb = (bf16_t*)carve((size_t)NH  * HD * D_MODEL * 2);
    bf16_t* wkb = (bf16_t*)carve((size_t)NKV * HD * D_MODEL * 2);
    bf16_t* wvb = (bf16_t*)carve((size_t)NKV * HD * D_MODEL * 2);
    bf16_t* wob = (bf16_t*)carve((size_t)D_MODEL * NH * HD * 2);
    float*  qf  = (float*) carve((size_t)S_LEN * NH  * HD * 4);
    float*  kf  = (float*) carve((size_t)S_LEN * NKV * HD * 4);
    float*  vf  = (float*) carve((size_t)S_LEN * NKV * HD * 4);
    bf16_t* qrb = (bf16_t*)carve((size_t)S_LEN * NH  * HD * 2);
    bf16_t* krb = (bf16_t*)carve((size_t)S_LEN * NKV * HD * 2);
    bf16_t* vtb = (bf16_t*)carve((size_t)NKV * HD * S_LEN * 2);
    bf16_t* ab  = (bf16_t*)carve((size_t)S_LEN * NH * HD * 2);

    // 1) fp32 -> bf16 conversions
    cvt_bf16_kernel<<<1024, 256, 0, stream>>>(x,  xb,  S_LEN * D_MODEL);
    cvt_bf16_kernel<<<1024, 256, 0, stream>>>(Wq, wqb, NH * HD * D_MODEL);
    cvt_bf16_kernel<<<1024, 256, 0, stream>>>(Wk, wkb, NKV * HD * D_MODEL);
    cvt_bf16_kernel<<<1024, 256, 0, stream>>>(Wv, wvb, NKV * HD * D_MODEL);
    cvt_bf16_kernel<<<1024, 256, 0, stream>>>(Wo, wob, D_MODEL * NH * HD);

    // 2) QKV projections (async-staged, double-buffered WMMA GEMM)
    gemm_bf16_kernel<<<dim3((NH * HD) / GBN, S_LEN / GBM), 256, 0, stream>>>(
        xb, wqb, qf, S_LEN, NH * HD, D_MODEL);
    gemm_bf16_kernel<<<dim3((NKV * HD) / GBN, S_LEN / GBM), 256, 0, stream>>>(
        xb, wkb, kf, S_LEN, NKV * HD, D_MODEL);
    gemm_bf16_kernel<<<dim3((NKV * HD) / GBN, S_LEN / GBM), 256, 0, stream>>>(
        xb, wvb, vf, S_LEN, NKV * HD, D_MODEL);

    // 3) RMS-norm + RoPE -> bf16
    rmsnorm_rope_kernel<<<S_LEN * NH, HD, 0, stream>>>(qf, cosT, sinT, qnw, qrb, NH);
    rmsnorm_rope_kernel<<<S_LEN * NKV, HD, 0, stream>>>(kf, cosT, sinT, knw, krb, NKV);

    // 4) V transpose to [NKV][HD][S] bf16
    vtrans_kernel<<<1024, 256, 0, stream>>>(vf, vtb);

    // 5) attention (TDM Q stage, LDS scores, exact top-40, softmax, WMMA)
    attn_kernel<<<dim3(S_LEN / 16, NH), 256, ATTN_LDS, stream>>>(qrb, krb, vtb, ab);

    // 6) output projection -> fp32 d_out
    gemm_bf16_kernel<<<dim3(D_MODEL / GBN, S_LEN / GBM), 256, 0, stream>>>(
        ab, wob, out, S_LEN, D_MODEL, NH * HD);
}
